// SINQQuantizedLinear_64330020159905
// MI455X (gfx1250) — compile-verified, matrix operand order checked by reading, not run on hardware
//
#include <hip/hip_runtime.h>

// Problem geometry (fixed by the reference)
#define MDIM 1024   // 4*256 flattened batch rows
#define NDIM 4096   // inner (reduction) dimension
#define KDIM 4096   // output features
#define NGRP 32     // NDIM / 128 groups

typedef __attribute__((ext_vector_type(16))) __bf16         v16bf;
typedef __attribute__((ext_vector_type(8)))  float          v8f;
typedef __attribute__((ext_vector_type(4)))  float          v4f;
typedef __attribute__((ext_vector_type(4)))  int            v4i;
typedef __attribute__((ext_vector_type(4)))  unsigned int   v4u;
typedef __attribute__((ext_vector_type(4)))  unsigned short v4s;

// float -> bf16 round-to-nearest-even (bit manipulation; no __bf16 arithmetic)
__device__ __forceinline__ unsigned short f2bf(float f) {
  unsigned u = __float_as_uint(f);
  u += 0x7FFFu + ((u >> 16) & 1u);
  return (unsigned short)(u >> 16);
}

struct Pair { v4u a; v4u b; };  // 32 bytes, trivially copyable

// Load one 16x32-bf16 fragment for this lane.
// p points at element (row, nb + 8*(lane>=16)); hi chunk is +16 elements (K+16).
__device__ __forceinline__ v16bf load_frag(const unsigned short* __restrict__ p) {
  Pair t;
  t.a = *(const v4u*)(p);        // vector elems 0..7  -> K = kbase .. kbase+7
  t.b = *(const v4u*)(p + 16);   // vector elems 8..15 -> K = kbase+16 .. kbase+23
  return __builtin_bit_cast(v16bf, t);
}

// ---------------------------------------------------------------------------
// Pass 1a: xb[m,n] = bf16( x[m,n] * mu1[n] )      (MDIM x NDIM, 8 MB)
// ---------------------------------------------------------------------------
__global__ __launch_bounds__(256) void sinq_prep_x(const float* __restrict__ x,
                                                   const float* __restrict__ mu1,
                                                   unsigned short* __restrict__ xb) {
  const int idx = (blockIdx.x * 256 + threadIdx.x) * 4;
  const int n   = idx & (NDIM - 1);
  const v4f xv = *(const v4f*)(x + idx);
  const v4f mv = *(const v4f*)(mu1 + n);
  v4s o;
  o.x = f2bf(xv.x * mv.x);
  o.y = f2bf(xv.y * mv.y);
  o.z = f2bf(xv.z * mv.z);
  o.w = f2bf(xv.w * mv.w);
  *(v4s*)(xb + idx) = o;
}

// ---------------------------------------------------------------------------
// Pass 1b: wb[k,n] = bf16( (Q[k,n] - zeros[k,g]) * scales[k,g] ),  g = n>>7
//          (KDIM x NDIM, 32 MB — lives in the 192 MB L2 for the GEMM pass)
// ---------------------------------------------------------------------------
__global__ __launch_bounds__(256) void sinq_prep_w(const int*   __restrict__ Wq,
                                                   const float* __restrict__ scales,
                                                   const float* __restrict__ zeros,
                                                   unsigned short* __restrict__ wb) {
  const int idx = (blockIdx.x * 256 + threadIdx.x) * 4;
  const int k   = idx >> 12;           // / NDIM
  const int n   = idx & (NDIM - 1);
  const int g   = n >> 7;              // group of 128 (4 consecutive n share g)
  const float s = scales[k * NGRP + g];
  const float z = zeros [k * NGRP + g];
  const float nzs = -z * s;            // w = q*s + (-z*s)
  const v4i q = *(const v4i*)(Wq + idx);
  v4s o;
  o.x = f2bf(fmaf((float)q.x, s, nzs));
  o.y = f2bf(fmaf((float)q.y, s, nzs));
  o.z = f2bf(fmaf((float)q.z, s, nzs));
  o.w = f2bf(fmaf((float)q.w, s, nzs));
  *(v4s*)(wb + idx) = o;
}

// ---------------------------------------------------------------------------
// Pass 2: out[m,k] = mu2[k] * ( xb[m,:] . wb[k,:] ) + bias[k]
// WG = 128 threads = 4 waves (2M x 2K); WG tile 128(M) x 128(K);
// wave tile 64 x 64 -> 16 accumulator tiles, 16 b128 loads per 16 WMMAs.
// v_wmma_f32_16x16x32_bf16, fragments loaded straight from global (L2-backed).
// ---------------------------------------------------------------------------
__global__ __launch_bounds__(128) void sinq_gemm_bf16(
    const unsigned short* __restrict__ xb,   // [MDIM, NDIM] bf16
    const unsigned short* __restrict__ wb,   // [KDIM, NDIM] bf16
    const float* __restrict__ mu2,           // [KDIM]
    const float* __restrict__ bias,          // [KDIM]
    float* __restrict__ out)                 // [MDIM, KDIM]
{
  const int lane = threadIdx.x & 31;
  const int wave = threadIdx.x >> 5;  // 0..3
  const int wm   = wave & 1;          // 2 M-waves
  const int wk   = wave >> 1;         // 2 K-waves
  const int m0   = blockIdx.y * 128 + wm * 64;
  const int k0   = blockIdx.x * 128 + wk * 64;
  const int half = lane >> 4;         // 0: K 0..7/16..23   1: K 8..15/24..31
  const int l16  = lane & 15;

  // Per-lane fragment base pointers (row-major, bf16 elements)
  const unsigned short* aBase[4];
  const unsigned short* bBase[4];
#pragma unroll
  for (int i = 0; i < 4; ++i) {
    aBase[i] = xb + (size_t)(m0 + i * 16 + l16) * NDIM + half * 8;
    bBase[i] = wb + (size_t)(k0 + i * 16 + l16) * NDIM + half * 8;
  }

  v8f acc[4][4] = {};

  for (int nb = 0; nb < NDIM; nb += 128) {
    // Pull the W' strips ~4KB ahead into near caches (WGP-scope prefetch);
    // once per 128-element strip, not per 32-step.
#pragma unroll
    for (int j = 0; j < 4; ++j)
      __builtin_prefetch(bBase[j] + nb + 2048, 0, 3);

#pragma unroll
    for (int t = 0; t < 4; ++t) {
      const int nn = nb + t * 32;
      v16bf a[4], b[4];
#pragma unroll
      for (int i = 0; i < 4; ++i) a[i] = load_frag(aBase[i] + nn);
#pragma unroll
      for (int j = 0; j < 4; ++j) b[j] = load_frag(bBase[j] + nn);

#pragma unroll
      for (int i = 0; i < 4; ++i)
#pragma unroll
        for (int j = 0; j < 4; ++j)
          acc[i][j] = __builtin_amdgcn_wmma_f32_16x16x32_bf16(
              false, a[i], false, b[j], (short)0, acc[i][j], false, false);
    }
  }

  // Epilogue: out[m,k] = mu2[k]*acc + bias[k]
  // C/D layout: VGPR r -> row (r + 8*half) within a 16-row tile; lane&15 -> column.
  float m2[4], bv[4];
#pragma unroll
  for (int j = 0; j < 4; ++j) {
    const int col = k0 + j * 16 + l16;
    m2[j] = mu2[col];
    bv[j] = bias[col];
  }
#pragma unroll
  for (int i = 0; i < 4; ++i) {
#pragma unroll
    for (int j = 0; j < 4; ++j) {
      const int col = k0 + j * 16 + l16;
#pragma unroll
      for (int r = 0; r < 8; ++r) {
        const int row = m0 + i * 16 + half * 8 + r;
        out[(size_t)row * KDIM + col] = fmaf(acc[i][j][r], m2[j], bv[j]);
      }
    }
  }
}

extern "C" void kernel_launch(void* const* d_in, const int* in_sizes, int n_in,
                              void* d_out, int out_size, void* d_ws, size_t ws_size,
                              hipStream_t stream) {
  (void)in_sizes; (void)n_in; (void)out_size; (void)ws_size;
  const float* x      = (const float*)d_in[0];  // [4,256,4096]
  const int*   Wq     = (const int*)  d_in[1];  // [4096,4096] int32 in [0,16)
  const float* scales = (const float*)d_in[2];  // [4096,32,1]
  const float* zeros  = (const float*)d_in[3];  // [4096,32,1]
  const float* mu1    = (const float*)d_in[4];  // [4096]
  const float* mu2    = (const float*)d_in[5];  // [4096]
  const float* bias   = (const float*)d_in[6];  // [4096]
  float* out = (float*)d_out;                   // [4,256,4096] f32

  unsigned short* xb = (unsigned short*)d_ws;                                    // 8 MB
  unsigned short* wb = (unsigned short*)((char*)d_ws + (size_t)MDIM * NDIM * 2); // 32 MB

  sinq_prep_x<<<(MDIM * NDIM) / (4 * 256), 256, 0, stream>>>(x, mu1, xb);
  sinq_prep_w<<<(KDIM * NDIM) / (4 * 256), 256, 0, stream>>>(Wq, scales, zeros, wb);

  dim3 grid(KDIM / 128, MDIM / 128);  // 32 x 8 = 256 workgroups
  sinq_gemm_bf16<<<grid, 128, 0, stream>>>(xb, wb, mu2, bias, out);
}
